// LlamaAttention_64974265254493
// MI455X (gfx1250) — matrix operand order repaired and checked by reading, not compiled, not run on hardware
//
#include <hip/hip_runtime.h>

// ---------------------------------------------------------------------------
// Types
// ---------------------------------------------------------------------------
typedef __bf16 bf16;
typedef __attribute__((ext_vector_type(16))) __bf16 v16bf;
typedef __attribute__((ext_vector_type(8)))  __bf16 v8bf;
typedef __attribute__((ext_vector_type(4)))  __bf16 v4bf;
typedef __attribute__((ext_vector_type(8)))  float  v8f;
typedef __attribute__((ext_vector_type(4)))  float  v4f;

#define B_    2
#define SEQ   2048
#define H_    4096
#define NH_   32
#define NKV_  8
#define HD_   128
#define QKV_N 6144
#define NTOK  (B_ * SEQ)   /* 4096 tokens */

// ---------------------------------------------------------------------------
// bf16 helpers (round-to-nearest-even)
// ---------------------------------------------------------------------------
__device__ __forceinline__ bf16 f2bf(float f) {
    unsigned u = __builtin_bit_cast(unsigned, f);
    u += 0x7fffu + ((u >> 16) & 1u);
    unsigned short h = (unsigned short)(u >> 16);
    return __builtin_bit_cast(bf16, h);
}
__device__ __forceinline__ float bf2f(bf16 b) {
    unsigned short h = __builtin_bit_cast(unsigned short, b);
    unsigned u = ((unsigned)h) << 16;
    return __builtin_bit_cast(float, u);
}
__device__ __forceinline__ v8f zero8() {
    v8f z;
#pragma unroll
    for (int i = 0; i < 8; ++i) z[i] = 0.0f;
    return z;
}

// ---------------------------------------------------------------------------
// CDNA5 async memory->LDS copy (GLOBAL_LOAD_ASYNC_TO_LDS_B128, ASYNCcnt).
// LDS operand is the 32-bit addrspace(3) pointer in one VGPR; global address
// is a 64-bit VGPR pair.
// ---------------------------------------------------------------------------
__device__ __forceinline__ void async_b128(const bf16* g, bf16* l) {
    __attribute__((address_space(3))) bf16* lp =
        (__attribute__((address_space(3))) bf16*)l;
    asm volatile("global_load_async_to_lds_b128 %0, %1, off"
                 :: "v"(lp), "v"(g) : "memory");
}
__device__ __forceinline__ void wait_async4() {
    asm volatile("s_wait_asynccnt 0x4" ::: "memory");
}
__device__ __forceinline__ void wait_async0() {
    asm volatile("s_wait_asynccnt 0x0" ::: "memory");
}

// ---------------------------------------------------------------------------
// WMMA fragment loaders (layouts per CDNA5 ISA 7.12.2, 16-bit, wave32)
// A 16x32 (MxK): lane<16 holds row lane, K chunks [0..7] and [16..23];
//                lane>=16 same row, K chunks [8..15] and [24..31].
// B 32x16 (KxN): lane holds column (lane&15), K chunks [0..7] and [16..23]
//                (lane<16) or [8..15] and [24..31] (lane>=16), same as A.
// ---------------------------------------------------------------------------
__device__ __forceinline__ v16bf load_frag_a(const bf16* base, int stride, int lane) {
    const bf16* p = base + (lane & 15) * stride + ((lane >> 4) << 3);
    v8bf lo = *(const v8bf*)(p);
    v8bf hi = *(const v8bf*)(p + 16);
    v16bf f;
#pragma unroll
    for (int i = 0; i < 8; ++i) { f[i] = lo[i]; f[i + 8] = hi[i]; }
    return f;
}
__device__ __forceinline__ v16bf load_frag_b(const bf16* base, int stride, int lane) {
    const bf16* p = base + (lane & 15) * stride + ((lane >> 4) << 3);
    v8bf lo = *(const v8bf*)(p);
    v8bf hi = *(const v8bf*)(p + 16);
    v16bf f;
#pragma unroll
    for (int i = 0; i < 8; ++i) { f[i] = lo[i]; f[i + 8] = hi[i]; }
    return f;
}
__device__ __forceinline__ v8f wmma_bf16(v16bf a, v16bf b, v8f c) {
    return __builtin_amdgcn_wmma_f32_16x16x32_bf16(false, a, false, b, (short)0, c,
                                                   false, false);
}

// half-wave (16-lane) reductions for softmax rows (C-layout: N striped on lanes)
__device__ __forceinline__ float rmax16(float v) {
#pragma unroll
    for (int m = 1; m < 16; m <<= 1) v = fmaxf(v, __shfl_xor(v, m, 32));
    return v;
}
__device__ __forceinline__ float rsum16(float v) {
#pragma unroll
    for (int m = 1; m < 16; m <<= 1) v += __shfl_xor(v, m, 32);
    return v;
}

// ---------------------------------------------------------------------------
// Kernel 1: fp32 -> bf16 conversion (vectorized x4)
// ---------------------------------------------------------------------------
__global__ void cvt_f32_bf16(const float* __restrict__ src, bf16* __restrict__ dst,
                             long n4) {
    long i = (long)blockIdx.x * blockDim.x + threadIdx.x;
    if (i >= n4) return;
    v4f x = *(const v4f*)(src + i * 4);
    v4bf y;
#pragma unroll
    for (int j = 0; j < 4; ++j) y[j] = f2bf(x[j]);
    *(v4bf*)(dst + i * 4) = y;
}

// ---------------------------------------------------------------------------
// Kernel 2/5: bf16 GEMM  C[M,N] = A[M,K] * W[N,K]^T
// 128x128 block tile, K-step 32, 256 threads = 8 waves, each wave 32x64.
// Double-buffered LDS tiles filled by async loads (ASYNCcnt), 1 barrier/step.
// OUTF32=0 -> bf16 C, OUTF32=1 -> fp32 C.
// ---------------------------------------------------------------------------
template <int OUTF32>
__global__ __launch_bounds__(256) void gemm_bf16(const bf16* __restrict__ A,
                                                 const bf16* __restrict__ W,
                                                 void* __restrict__ Cp, int M, int N,
                                                 int K) {
    __shared__ bf16 lA[2][128 * 40];  // 32 cols + 8 pad (80B row stride, 16B aligned)
    __shared__ bf16 lB[2][128 * 40];

    const int tm = blockIdx.y * 128, tn = blockIdx.x * 128;
    const int wave = threadIdx.x >> 5, lane = threadIdx.x & 31;
    const int wm = (wave >> 1) * 32, wn = (wave & 1) * 64;

    v8f acc[2][4];
#pragma unroll
    for (int i = 0; i < 2; ++i)
#pragma unroll
        for (int j = 0; j < 4; ++j) acc[i][j] = zero8();

    // issue one K-tile (32 deep) of async loads: 4 async instrs per wave
    auto issue = [&](int buf, int k0) {
        for (int c = threadIdx.x; c < 512; c += 256) {
            int r = c >> 2, cc = (c & 3) << 3;
            async_b128(A + (long)(tm + r) * K + k0 + cc, &lA[buf][r * 40 + cc]);
        }
        for (int c = threadIdx.x; c < 512; c += 256) {
            int r = c >> 2, cc = (c & 3) << 3;
            async_b128(W + (long)(tn + r) * K + k0 + cc, &lB[buf][r * 40 + cc]);
        }
    };

    issue(0, 0);
    for (int k0 = 0; k0 < K; k0 += 32) {
        const int buf = (k0 >> 5) & 1;
        if (k0 + 32 < K) {
            issue(buf ^ 1, k0 + 32);
            wait_async4();   // current tile's 4 asyncs retired; next 4 in flight
        } else {
            wait_async0();
        }
        __syncthreads();

        v16bf af[2], bw[4];
        af[0] = load_frag_a(&lA[buf][(wm + 0) * 40], 40, lane);
        af[1] = load_frag_a(&lA[buf][(wm + 16) * 40], 40, lane);
#pragma unroll
        for (int j = 0; j < 4; ++j)
            bw[j] = load_frag_b(&lB[buf][(wn + j * 16) * 40], 40, lane);
#pragma unroll
        for (int i = 0; i < 2; ++i)
#pragma unroll
            for (int j = 0; j < 4; ++j) acc[i][j] = wmma_bf16(af[i], bw[j], acc[i][j]);
        // no trailing barrier: next loads target the other buffer; reuse of this
        // buffer happens 2 iterations later, beyond the next barrier.
    }

    // C layout: VGPR e -> row e (lanes 0-15) / e+8 (lanes 16-31); N = lane&15
    const int laneR = lane & 15, mOff = (lane >> 4) << 3;
#pragma unroll
    for (int i = 0; i < 2; ++i)
#pragma unroll
        for (int j = 0; j < 4; ++j) {
            int colg = tn + wn + j * 16 + laneR;
#pragma unroll
            for (int e = 0; e < 8; ++e) {
                long rowg = tm + wm + i * 16 + mOff + e;
                if (OUTF32)
                    ((float*)Cp)[rowg * (long)N + colg] = acc[i][j][e];
                else
                    ((bf16*)Cp)[rowg * (long)N + colg] = f2bf(acc[i][j][e]);
            }
        }
}

// ---------------------------------------------------------------------------
// Kernel 3: RoPE + reorder to head-major [b][h][s][d]   (for Q and K)
// ---------------------------------------------------------------------------
__global__ void rope_reorder(const bf16* __restrict__ qkv, bf16* __restrict__ dst,
                             int nHeads, int colBase) {
    long i = (long)blockIdx.x * blockDim.x + threadIdx.x;
    long total = (long)NTOK * nHeads * 64;
    if (i >= total) return;
    int d = (int)(i & 63);
    int h = (int)((i >> 6) % nHeads);
    long t = i / (64L * nHeads);        // token = b*SEQ + s
    int s = (int)(t & (SEQ - 1));
    int b = (int)(t >> 11);
    float ang = (float)s * __powf(10000.0f, -(float)(2 * d) / 128.0f);
    float sn, cs;
    __sincosf(ang, &sn, &cs);
    const bf16* row = qkv + t * QKV_N + colBase + h * HD_;
    float x1 = bf2f(row[d]);
    float x2 = bf2f(row[d + 64]);
    bf16* drow = dst + (((long)b * nHeads + h) * SEQ + s) * HD_;
    drow[d]      = f2bf(x1 * cs - x2 * sn);
    drow[d + 64] = f2bf(x1 * sn + x2 * cs);
}

// Kernel 3b: V transpose to [b][kv][d][s] (K-contiguous for PV B-operand)
__global__ void v_transpose(const bf16* __restrict__ qkv, bf16* __restrict__ vt) {
    long i = (long)blockIdx.x * blockDim.x + threadIdx.x;
    long total = (long)NTOK * NKV_ * HD_;
    if (i >= total) return;
    int d = (int)(i & 127);
    int kv = (int)((i >> 7) & 7);
    long t = i >> 10;
    int s = (int)(t & (SEQ - 1));
    int b = (int)(t >> 11);
    vt[(((long)b * NKV_ + kv) * HD_ + d) * SEQ + s] =
        qkv[t * QKV_N + (H_ + NKV_ * HD_) + kv * HD_ + d];
}

// ---------------------------------------------------------------------------
// Kernel 4: causal flash attention.
// Grid: (S/128, B*NH). 256 threads = 8 waves; wave w owns q rows [w*16, w*16+16).
// K/V staged in double-buffered LDS k-blocks via async loads; online softmax;
// P recirculated via per-wave LDS scratch into A-fragment layout.
// ---------------------------------------------------------------------------
__global__ __launch_bounds__(256) void flash_attn(const bf16* __restrict__ Q,
                                                  const bf16* __restrict__ Kc,
                                                  const bf16* __restrict__ Vt,
                                                  bf16* __restrict__ Ao) {
    const int bh = blockIdx.y;           // b*NH + h
    const int b = bh >> 5;
    const int h = bh & 31;
    const int kv = h >> 2;               // NH/NKV = 4
    const int qt = blockIdx.x;
    const int wave = threadIdx.x >> 5, lane = threadIdx.x & 31;
    const int laneR = lane & 15;
    const int mOff = (lane >> 4) << 3;

    __shared__ bf16 lK[2][32 * 136];     // [k][d] rows of K block (272B stride)
    __shared__ bf16 lV[2][128 * 40];     // [d][k] rows of V^T block
    __shared__ bf16 lP[8][16 * 40];      // per-wave P scratch 16x32

    const int q0 = qt * 128 + wave * 16;
    const bf16* qp = Q + ((long)bh * SEQ + q0) * HD_;

    v16bf qf[4];                         // Q A-fragments over d chunks of 32
#pragma unroll
    for (int dc = 0; dc < 4; ++dc) {
        const bf16* p = qp + laneR * HD_ + dc * 32 + mOff;
        v8bf lo = *(const v8bf*)(p);
        v8bf hi = *(const v8bf*)(p + 16);
        v16bf f;
#pragma unroll
        for (int i = 0; i < 8; ++i) { f[i] = lo[i]; f[i + 8] = hi[i]; }
        qf[dc] = f;
    }

    v8f o[8];
#pragma unroll
    for (int t = 0; t < 8; ++t) o[t] = zero8();
    float m_[8], l_[8];
#pragma unroll
    for (int j = 0; j < 8; ++j) { m_[j] = -3.0e38f; l_[j] = 0.0f; }

    const int qlast = q0 + 15;
    const int nkb = (qt + 1) * 4;        // causal: k < (qt+1)*128
    const bf16* kbase = Kc + ((long)b * NKV_ + kv) * SEQ * (long)HD_;
    const bf16* vbase = Vt + ((long)b * NKV_ + kv) * HD_ * (long)SEQ;
    const float SCALE = 0.08838834764831845f;  // 1/sqrt(128)

    // issue one 32-wide k-block: 4 async instrs per wave (2 K + 2 V)
    auto issueKV = [&](int buf, int kst) {
        for (int c = threadIdx.x; c < 512; c += 256) {  // K: 32 rows x 128 d
            int r = c >> 4, cc = (c & 15) << 3;
            async_b128(kbase + (long)(kst + r) * HD_ + cc, &lK[buf][r * 136 + cc]);
        }
        for (int c = threadIdx.x; c < 512; c += 256) {  // V^T: 128 d x 32 k
            int r = c >> 2, cc = (c & 3) << 3;
            async_b128(vbase + (long)r * SEQ + kst + cc, &lV[buf][r * 40 + cc]);
        }
    };

    issueKV(0, 0);
    for (int kb = 0; kb < nkb; ++kb) {
        const int kst = kb * 32;
        const int buf = kb & 1;
        if (kb + 1 < nkb) {
            issueKV(buf ^ 1, kst + 32);
            wait_async4();
        } else {
            wait_async0();
        }
        __syncthreads();

        if (kst <= qlast) {
            v8f s0 = zero8(), s1 = zero8();
#pragma unroll
            for (int dc = 0; dc < 4; ++dc) {
                v16bf k0f = load_frag_b(&lK[buf][0 * 136 + dc * 32], 136, lane);
                v16bf k1f = load_frag_b(&lK[buf][16 * 136 + dc * 32], 136, lane);
                s0 = wmma_bf16(qf[dc], k0f, s0);
                s1 = wmma_bf16(qf[dc], k1f, s1);
            }
            float alpha[8];
#pragma unroll
            for (int j = 0; j < 8; ++j) {
                int qrow = q0 + j + mOff;
                float a = s0[j] * SCALE;
                if (kst + laneR > qrow) a = -3.0e38f;
                float bb = s1[j] * SCALE;
                if (kst + 16 + laneR > qrow) bb = -3.0e38f;
                float mn = fmaxf(m_[j], rmax16(fmaxf(a, bb)));
                float p0 = __expf(a - mn);
                float p1 = __expf(bb - mn);
                float al = __expf(m_[j] - mn);
                m_[j] = mn;
                l_[j] = l_[j] * al + rsum16(p0 + p1);
                alpha[j] = al;
                lP[wave][(j + mOff) * 40 + laneR] = f2bf(p0);
                lP[wave][(j + mOff) * 40 + 16 + laneR] = f2bf(p1);
            }
#pragma unroll
            for (int t = 0; t < 8; ++t)
#pragma unroll
                for (int j = 0; j < 8; ++j) o[t][j] *= alpha[j];

            __asm volatile("s_wait_dscnt 0x0" ::: "memory");  // P RAW through LDS

            v16bf pf = load_frag_a(&lP[wave][0], 40, lane);
#pragma unroll
            for (int t = 0; t < 8; ++t) {
                v16bf vf = load_frag_b(&lV[buf][t * 16 * 40], 40, lane);
                o[t] = wmma_bf16(pf, vf, o[t]);
            }
        }
        // no trailing barrier: next async loads target the other buffer.
    }

    // epilogue: O/l -> attn buffer [token][h*128+d], bf16
    bf16* ab = Ao + ((long)b * SEQ + q0) * H_ + h * HD_;
#pragma unroll
    for (int j = 0; j < 8; ++j) {
        float inv = 1.0f / l_[j];
#pragma unroll
        for (int t = 0; t < 8; ++t)
            ab[(long)(j + mOff) * H_ + t * 16 + laneR] = f2bf(o[t][j] * inv);
    }
}

// ---------------------------------------------------------------------------
// Host launcher
// ---------------------------------------------------------------------------
extern "C" void kernel_launch(void* const* d_in, const int* in_sizes, int n_in,
                              void* d_out, int out_size, void* d_ws, size_t ws_size,
                              hipStream_t stream) {
    const float* hs = (const float*)d_in[0];
    const float* wq = (const float*)d_in[1];
    const float* wk = (const float*)d_in[2];
    const float* wv = (const float*)d_in[3];
    const float* wo = (const float*)d_in[4];

    // workspace carve-up (~176 MB with reuse)
    char* w = (char*)d_ws;
    bf16* hsb   = (bf16*)w; w += (size_t)NTOK * H_ * 2;       // 32MB (reused for wo)
    bf16* wqkvb = (bf16*)w; w += (size_t)QKV_N * H_ * 2;      // 48MB
    bf16* qkvb  = (bf16*)w; w += (size_t)NTOK * QKV_N * 2;    // 48MB (reused for attn)
    bf16* qro   = (bf16*)w; w += (size_t)B_ * NH_ * SEQ * HD_ * 2;   // 32MB
    bf16* kro   = (bf16*)w; w += (size_t)B_ * NKV_ * SEQ * HD_ * 2;  // 8MB
    bf16* vtb   = (bf16*)w; w += (size_t)B_ * NKV_ * SEQ * HD_ * 2;  // 8MB
    bf16* wob   = hsb;    // safe: wo conversion enqueued after QKV GEMM
    bf16* attnb = qkvb;   // safe: flash enqueued after rope/vtrans consume qkvb

    // 1) convert activations + packed QKV weights to bf16
    {
        long n4 = (long)NTOK * H_ / 4;
        cvt_f32_bf16<<<(unsigned)((n4 + 255) / 256), 256, 0, stream>>>(hs, hsb, n4);
        n4 = (long)H_ * H_ / 4;   // wq rows 0..4095
        cvt_f32_bf16<<<(unsigned)((n4 + 255) / 256), 256, 0, stream>>>(wq, wqkvb, n4);
        n4 = (long)NKV_ * HD_ * H_ / 4;  // wk rows 4096..5119
        cvt_f32_bf16<<<(unsigned)((n4 + 255) / 256), 256, 0, stream>>>(
            wk, wqkvb + (size_t)H_ * H_, n4);
        cvt_f32_bf16<<<(unsigned)((n4 + 255) / 256), 256, 0, stream>>>(
            wv, wqkvb + (size_t)(H_ + NKV_ * HD_) * H_, n4);  // wv rows 5120..6143
    }

    // 2) fused QKV projection: [4096 x 4096] @ [6144 x 4096]^T -> bf16
    gemm_bf16<0><<<dim3(QKV_N / 128, NTOK / 128), 256, 0, stream>>>(
        hsb, wqkvb, qkvb, NTOK, QKV_N, H_);

    // convert wo now (reuses hsb region; ordered after GEMM on stream)
    {
        long n4 = (long)H_ * H_ / 4;
        cvt_f32_bf16<<<(unsigned)((n4 + 255) / 256), 256, 0, stream>>>(wo, wob, n4);
    }

    // 3) RoPE + reorder Q, K; transpose V
    {
        long nq = (long)NTOK * NH_ * 64;
        rope_reorder<<<(unsigned)((nq + 255) / 256), 256, 0, stream>>>(qkvb, qro, NH_, 0);
        long nk = (long)NTOK * NKV_ * 64;
        rope_reorder<<<(unsigned)((nk + 255) / 256), 256, 0, stream>>>(qkvb, kro, NKV_, H_);
        long nv = (long)NTOK * NKV_ * HD_;
        v_transpose<<<(unsigned)((nv + 255) / 256), 256, 0, stream>>>(qkvb, vtb);
    }

    // 4) causal flash attention -> attn bf16 [token][4096]
    flash_attn<<<dim3(SEQ / 128, B_ * NH_), 256, 0, stream>>>(qro, kro, vtb, attnb);

    // 5) output projection: attn @ wo^T -> fp32 d_out
    gemm_bf16<1><<<dim3(H_ / 128, NTOK / 128), 256, 0, stream>>>(
        attnb, wob, d_out, NTOK, H_, H_);
}
